// KNRM_37409165148283
// MI455X (gfx1250) — matrix operand order repaired and checked by reading, not compiled, not run on hardware
//
#include <hip/hip_runtime.h>
#include <hip/hip_bf16.h>

// ---------------------------------------------------------------------------
// KNRM fused kernel for gfx1250 (MI455X).
//   knrm_main (grid 512x2, 128 thr = 4 waves):
//     - gather + L2-normalize query rows -> f16 LDS A tile
//     - per wave: double-buffered ASYNC global->LDS staging of raw f32 doc
//       rows (GLOBAL_LOAD_ASYNC_TO_LDS_B128 / s_wait_asynccnt), normalize to
//       f16 LDS B tile, 4x chained v_wmma_f32_16x16x32_f16 cosine tile,
//       fused 21-Gaussian-kernel pooling on the WMMA accumulators
//     - half-wave ds_bpermute reductions + log1p -> k[pair][b][21]
//   knrm_mlp: 21->10->5->1 MLP + sigmoid(s1 - s2).
// Embedding table (51 MB) is L2-resident (192 MB L2), so the fused gather is
// cheap; async staging hides the remaining L2 latency behind exp/WMMA work.
// ---------------------------------------------------------------------------

typedef __attribute__((ext_vector_type(16))) _Float16 v16h;
typedef __attribute__((ext_vector_type(8)))  _Float16 v8h;
typedef __attribute__((ext_vector_type(8)))  float    v8f;
typedef __attribute__((ext_vector_type(4)))  int      v4i;

#define B_SZ   512
#define QL     16
#define DL     512
#define DIM    128
#define KN     21
#define NTILES (DL / 16)        // 32
#define WAVES  4
#define TPW    (NTILES / WAVES) // 8 tiles per wave

// ---- CDNA5 async global->LDS path (guarded: falls back to sync loads) ----
#if defined(__has_builtin)
#  if __has_builtin(__builtin_amdgcn_global_load_async_to_lds_b128) && \
      __has_builtin(__builtin_amdgcn_s_wait_asynccnt)
#    define USE_ASYNC 1
#  endif
#endif
#ifndef USE_ASYNC
#  define USE_ASYNC 0
#endif

#if USE_ASYNC
// builtin signature (from clang diagnostic): (AS1 v4i*, AS3 v4i*, imm, imm)
typedef __attribute__((address_space(1))) v4i* gas_ptr;
typedef __attribute__((address_space(3))) v4i* las_ptr;
__device__ __forceinline__ void cp16_async(const float* g, float* l) {
    // per-lane 16B copy, ASYNCcnt-tracked, lowers to global_load_async_to_lds_b128
    __builtin_amdgcn_global_load_async_to_lds_b128(
        (gas_ptr)(void*)g, (las_ptr)(void*)l, 0, 0);
}
#endif

__device__ __constant__ float MUS[KN] = {
    -0.95f, -0.85f, -0.75f, -0.65f, -0.55f, -0.45f, -0.35f, -0.25f, -0.15f, -0.05f,
     0.05f,  0.15f,  0.25f,  0.35f,  0.45f,  0.55f,  0.65f,  0.75f,  0.85f,  0.95f,
     1.00f };
// 0.5 / sigma^2 : sigma=0.1 -> 50 ; exact sigma=0.001 -> 5e5
__device__ __constant__ float CEXP[KN] = {
    50.f, 50.f, 50.f, 50.f, 50.f, 50.f, 50.f, 50.f, 50.f, 50.f,
    50.f, 50.f, 50.f, 50.f, 50.f, 50.f, 50.f, 50.f, 50.f, 50.f,
    5.0e5f };

// cross-lane XOR shuffle via ds_bpermute (wave32-safe)
__device__ __forceinline__ float xor_f32(float v, int m) {
    int lane = (int)(threadIdx.x & 31);
    int src  = (lane ^ m) << 2;  // byte address
    return __int_as_float(__builtin_amdgcn_ds_bpermute(src, __float_as_int(v)));
}

__global__ __launch_bounds__(128) void knrm_main(
    const int* __restrict__ q1, const int* __restrict__ d1,
    const int* __restrict__ q2, const int* __restrict__ d2,
    const float* __restrict__ emb,
    float* __restrict__ kout)   // [2][B][KN]
{
    const int b    = blockIdx.x;
    const int pair = blockIdx.y;
    const int tid  = (int)threadIdx.x;
    const int wave = tid >> 5;
    const int lane = tid & 31;

    const int* __restrict__ q = pair ? q2 : q1;
    const int* __restrict__ d = pair ? d2 : d1;

    __shared__ __align__(16) _Float16 As[QL][DIM + 8];          // 272B row stride
    __shared__ __align__(16) _Float16 Bs[WAVES][16][DIM + 8];   // wave-private f16 doc tiles
#if USE_ASYNC
    // chunk-major f32 staging: chunk j (16B per lane) at [j*32 + lane] -> 2-way banked reads
    __shared__ __align__(16) float stageF[WAVES][2][16 * DIM];  // 8KB per buffer
#endif
    __shared__ float rnorm[QL];
    __shared__ float kpart[WAVES][KN][16];

    // ---------------- stage A: 16 query rows, gather + L2 normalize -> f16 LDS
    if (tid < QL) rnorm[tid] = 0.f;
    __syncthreads();
    {
        const int row = tid >> 3;          // 16 rows x 8 threads
        const int seg = (tid & 7) * 16;    // 16 floats per thread
        const int tok = q[b * QL + row];
        const float4* s4 = (const float4*)(emb + (size_t)tok * DIM + seg);
        float vbuf[16];
        #pragma unroll
        for (int j = 0; j < 4; ++j) {
            float4 v = s4[j];
            vbuf[4*j+0] = v.x; vbuf[4*j+1] = v.y; vbuf[4*j+2] = v.z; vbuf[4*j+3] = v.w;
        }
        float ss = 0.f;
        #pragma unroll
        for (int j = 0; j < 16; ++j) ss = fmaf(vbuf[j], vbuf[j], ss);
        atomicAdd(&rnorm[row], ss);
        __syncthreads();
        const float sc = 1.0f / fmaxf(sqrtf(rnorm[row]), 1e-12f);
        #pragma unroll
        for (int j = 0; j < 16; ++j) As[row][seg + j] = (_Float16)(vbuf[j] * sc);
    }
    __syncthreads();

    // ---------------- A fragments (CDNA5 16-bit A layout: two contiguous
    // 8-half groups per lane at K = ks*32 + 8*hi and +16)
    union AF { v16h v; v8h h[2]; };
    AF afrag[4];
    {
        const int am  = lane & 15;
        const int ahi = lane >> 4;
        #pragma unroll
        for (int ks = 0; ks < 4; ++ks) {
            const _Float16* base = &As[am][ks * 32 + 8 * ahi];
            afrag[ks].h[0] = *(const v8h*)(base);
            afrag[ks].h[1] = *(const v8h*)(base + 16);
        }
    }

    // ---------------- per-lane pooling accumulators (21 kernels x 8 C rows)
    float acc[KN][8];
    #pragma unroll
    for (int kk = 0; kk < KN; ++kk)
        #pragma unroll
        for (int r = 0; r < 8; ++r) acc[kk][r] = 0.f;

    const int rp   = lane & 15;   // doc row within tile (lanes rp, rp+16 share it)
    const int hsel = lane >> 4;   // which 64-float half of the row this lane handles

    // shared tail: WMMA on the staged f16 tile + fused Gaussian pooling
    auto mma_pool = [&]() {
        __syncthreads();   // uniform call count across waves; publishes Bs writes
        v8f c = {};
        #pragma unroll
        for (int ks = 0; ks < 4; ++ks) {
            union BF { v16h v; v8h h[2]; } bf;
            // CDNA5 16-bit B layout: lane = column N=lane%16, two contiguous
            // 8-half groups at K = ks*32 + 8*(lane/16) and +16
            const _Float16* bbase = &Bs[wave][lane & 15][ks * 32 + 8 * (lane >> 4)];
            bf.h[0] = *(const v8h*)(bbase);
            bf.h[1] = *(const v8h*)(bbase + 16);
            c = __builtin_amdgcn_wmma_f32_16x16x32_f16(
                    false, afrag[ks].v, false, bf.v, (short)0, c, false, false);
        }
        #pragma unroll
        for (int kk = 0; kk < KN; ++kk) {
            const float mu = MUS[kk];
            const float cc = CEXP[kk];
            #pragma unroll
            for (int r = 0; r < 8; ++r) {
                const float t = c[r] - mu;
                acc[kk][r] += __expf(-t * t * cc);
            }
        }
    };

#if USE_ASYNC
    float* const stg0 = &stageF[wave][0][0];
    float* const stg1 = &stageF[wave][1][0];

    // issue one doc tile: 16 async b128 copies; chunk-major LDS layout
    auto issue_tile = [&](int tile, float* stg) {
        const int tok = d[b * DL + tile * 16 + rp];
        const float* g = emb + (size_t)tok * DIM + hsel * 64;
        float* l = stg + lane * 4;          // 16B slot per lane within each chunk row
        #pragma unroll
        for (int j = 0; j < 16; ++j)
            cp16_async(g + j * 4, l + j * 128);   // chunk row stride = 512B
    };

    // consume a staged tile: normalize rows, write f16 Bs tile, then mma+pool
    auto consume = [&](const float* stg) {
        const float* l = stg + lane * 4;
        float ss = 0.f;
        #pragma unroll
        for (int j = 0; j < 16; ++j) {
            float4 v = *(const float4*)(l + j * 128);
            ss = fmaf(v.x, v.x, fmaf(v.y, v.y, fmaf(v.z, v.z, fmaf(v.w, v.w, ss))));
        }
        ss += xor_f32(ss, 16);
        const float sc = 1.0f / fmaxf(sqrtf(ss), 1e-12f);
        _Float16* bd = &Bs[wave][rp][hsel * 64];
        #pragma unroll
        for (int j = 0; j < 16; ++j) {
            float4 v = *(const float4*)(l + j * 128);
            bd[4*j+0] = (_Float16)(v.x * sc);
            bd[4*j+1] = (_Float16)(v.y * sc);
            bd[4*j+2] = (_Float16)(v.z * sc);
            bd[4*j+3] = (_Float16)(v.w * sc);
        }
        mma_pool();
    };

    issue_tile(wave, stg0);                         // prologue: tile 0 -> buf0
    for (int ti = 0; ti < TPW - 1; ++ti) {
        issue_tile(wave + (ti + 1) * WAVES, (ti & 1) ? stg0 : stg1); // next -> other buf
        __builtin_amdgcn_s_wait_asynccnt(16);       // current tile's 16 copies done
        consume((ti & 1) ? stg1 : stg0);
    }
    __builtin_amdgcn_s_wait_asynccnt(0);            // last tile fully landed
    consume(stg1);                                  // tile 7 lives in buf1
#else
    // -------- fallback: synchronous gather (round-1 path) with prefetch
    for (int ti = 0; ti < TPW; ++ti) {
        const int tile = wave + ti * WAVES;
        const int tok = d[b * DL + tile * 16 + rp];
        const float4* s4 = (const float4*)(emb + (size_t)tok * DIM + hsel * 64);
        if (ti < TPW - 1) {
            const int ntok = d[b * DL + (tile + WAVES) * 16 + rp];
            __builtin_prefetch(emb + (size_t)ntok * DIM + hsel * 64, 0, 0);
        }
        float ss = 0.f;
        #pragma unroll
        for (int j = 0; j < 16; ++j) {
            float4 v = s4[j];
            ss = fmaf(v.x, v.x, fmaf(v.y, v.y, fmaf(v.z, v.z, fmaf(v.w, v.w, ss))));
        }
        ss += xor_f32(ss, 16);
        const float sc = 1.0f / fmaxf(sqrtf(ss), 1e-12f);
        _Float16* bd = &Bs[wave][rp][hsel * 64];
        #pragma unroll
        for (int j = 0; j < 16; ++j) {
            float4 v = s4[j];
            bd[4*j+0] = (_Float16)(v.x * sc);
            bd[4*j+1] = (_Float16)(v.y * sc);
            bd[4*j+2] = (_Float16)(v.z * sc);
            bd[4*j+3] = (_Float16)(v.w * sc);
        }
        mma_pool();
    }
#endif

    // ---------------- reduce over N within half-waves; lanes 0 / 16 hold row sums
    #pragma unroll
    for (int kk = 0; kk < KN; ++kk) {
        #pragma unroll
        for (int r = 0; r < 8; ++r) {
            float s = acc[kk][r];
            s += xor_f32(s, 1);
            s += xor_f32(s, 2);
            s += xor_f32(s, 4);
            s += xor_f32(s, 8);
            if ((lane & 15) == 0)           // lane 0 -> M=r ; lane 16 -> M=r+8
                kpart[wave][kk][r + 8 * (lane >> 4)] = s;
        }
    }
    __syncthreads();

    // ---------------- combine waves, log1p over DL-sum, sum over 16 query rows
    if (tid < KN) {
        float ksum = 0.f;
        #pragma unroll
        for (int m = 0; m < 16; ++m) {
            const float tot = kpart[0][tid][m] + kpart[1][tid][m] +
                              kpart[2][tid][m] + kpart[3][tid][m];
            ksum += log1pf(tot);
        }
        kout[((size_t)pair * B_SZ + b) * KN + tid] = ksum;
    }
}

// ---------------------------------------------------------------------------
__global__ __launch_bounds__(128) void knrm_mlp(
    const float* __restrict__ kbuf,
    const float* __restrict__ W1, const float* __restrict__ bb1,
    const float* __restrict__ W2, const float* __restrict__ bb2,
    const float* __restrict__ W3, const float* __restrict__ bb3,
    float* __restrict__ out, int B)
{
    const int b = blockIdx.x * blockDim.x + (int)threadIdx.x;
    if (b >= B) return;
    float s[2];
    #pragma unroll
    for (int p = 0; p < 2; ++p) {
        const float* k = kbuf + ((size_t)p * B + b) * KN;
        float h1[10];
        #pragma unroll
        for (int j = 0; j < 10; ++j) {
            float a = bb1[j];
            #pragma unroll
            for (int i = 0; i < KN; ++i) a = fmaf(k[i], W1[i * 10 + j], a);
            h1[j] = fmaxf(a, 0.f);
        }
        float h2[5];
        #pragma unroll
        for (int j = 0; j < 5; ++j) {
            float a = bb2[j];
            #pragma unroll
            for (int i = 0; i < 10; ++i) a = fmaf(h1[i], W2[i * 5 + j], a);
            h2[j] = fmaxf(a, 0.f);
        }
        float o = bb3[0];
        #pragma unroll
        for (int i = 0; i < 5; ++i) o = fmaf(h2[i], W3[i], o);
        s[p] = o;
    }
    const float z = s[0] - s[1];
    out[b] = 1.0f / (1.0f + __expf(-z));
}

// ---------------------------------------------------------------------------
extern "C" void kernel_launch(void* const* d_in, const int* in_sizes, int n_in,
                              void* d_out, int out_size, void* d_ws, size_t ws_size,
                              hipStream_t stream)
{
    const int*   q1  = (const int*)  d_in[0];
    const int*   d1  = (const int*)  d_in[1];
    const int*   q2  = (const int*)  d_in[2];
    const int*   d2  = (const int*)  d_in[3];
    const float* emb = (const float*)d_in[4];
    const float* W1  = (const float*)d_in[5];
    const float* b1  = (const float*)d_in[6];
    const float* W2  = (const float*)d_in[7];
    const float* b2  = (const float*)d_in[8];
    const float* W3  = (const float*)d_in[9];
    const float* b3  = (const float*)d_in[10];

    float* kbuf = (float*)d_ws;           // [2][512][21] floats = 84 KB scratch

    knrm_main<<<dim3(B_SZ, 2), 128, 0, stream>>>(q1, d1, q2, d2, emb, kbuf);
    knrm_mlp<<<(B_SZ + 127) / 128, 128, 0, stream>>>(kbuf, W1, b1, W2, b2, W3, b3,
                                                     (float*)d_out, B_SZ);
}
